// DrugResponseModel_67302137528741
// MI455X (gfx1250) — compile-verified, matrix operand order checked by reading, not run on hardware
//
#include <hip/hip_runtime.h>
#include <cstdint>

// ---------------------------------------------------------------------------
// DrugResponseModel for MI455X (gfx1250, wave32, WMMA, async-to-LDS).
// bf16 activations in HBM (memory-bound model -> halve traffic). ONE GEMM
// kernel, single code path: every call site has M%128==0, K%32==0, ld%8==0,
// so both A and B tiles are staged exclusively with per-lane
// GLOBAL_LOAD_ASYNC_TO_LDS_B128 (ASYNCcnt) into double-buffered LDS.
// Attention: head-padded Q/K weights (K=32 per-head scores) and a transposed-V
// epilogue (c_trans) remove all non-async staging paths.
// Workspace: ~1.45 GB (700MB f32 scatter acc + 2x352MB bf16 node ping-pong,
// with attention scores / probs / CNN intermediates aliased into them).
// ---------------------------------------------------------------------------

#define NODES 131072
#define EDGES 524288
#define BATCH 2048
#define LCELL 955
#define FD    334
#define DIM   128
#define HEADS 8

typedef __bf16 bf16_t;
typedef bf16_t v16bf __attribute__((ext_vector_type(16)));
typedef bf16_t v8bf  __attribute__((ext_vector_type(8)));
typedef float  v8f   __attribute__((ext_vector_type(8)));
typedef unsigned short ushort_t;

__device__ __forceinline__ ushort_t f2bf(float f) {
  unsigned u = __float_as_uint(f);
  u += 0x7FFFu + ((u >> 16) & 1u);           // round-to-nearest-even
  return (ushort_t)(u >> 16);
}
__device__ __forceinline__ float bf2f(ushort_t u) {
  return __uint_as_float((unsigned)u << 16);
}

// ---------------------------------------------------------------------------
// GEMM: C[M,N] = act(A[M,K]*B^T + bias); A [M,K] bf16, B [N,K] bf16 (both in
// HBM, 16B-aligned rows), C f32 or bf16, optionally transposed (C[N,M]).
// Preconditions (all call sites): M%128==0, K%32==0, lda%8==0, ldb%8==0.
// Epilogue writes cols [0,Nstore): computed value for col<N, zero pad after.
// 256 thr = 8 waves (4Mx2N); wave: 32x64 = 2x4 WMMA 16x16x32 bf16 tiles.
// ---------------------------------------------------------------------------
#define BM 128
#define BN 128
#define BK 32
#define LDT 48   // LDS row stride in bf16 elems: 96B (data 64B + pad)

__global__ __launch_bounds__(256)
void gemm_bf16_wmma(const ushort_t* __restrict__ A, long lda,
                    const ushort_t* __restrict__ Bm, long ldb,
                    const float* __restrict__ bias,
                    void* __restrict__ Cv, long ldc, int c_bf16, int c_trans,
                    int M, int N, int Nstore, int K, int relu)
{
  __shared__ __align__(16) ushort_t sA[2][BM * LDT];
  __shared__ __align__(16) ushort_t sB[2][BN * LDT];

  const int tid  = threadIdx.x;
  const int lane = tid & 31;
  const int wave = tid >> 5;
  const int wm = wave & 3;                 // 4 wave-rows * 32
  const int wn = wave >> 2;                // 2 wave-cols * 64
  const int rowBase = blockIdx.y * BM;
  const int colBase = blockIdx.x * BN;
  const int lr = lane & 15;
  const int lh = lane >> 4;

  // per-lane async copy of a 128x32 bf16 tile: 512 16B chunks, 2 per thread
  auto async_tile = [&](ushort_t* ldsTile, const ushort_t* gbase, long ld, int base) {
    const unsigned lbase = (unsigned)(size_t)ldsTile;
    {
      const int row = tid >> 2, part = tid & 3;
      const ushort_t* gp = gbase + (long)(base + row) * ld + part * 8;
      const unsigned lds = lbase + row * (LDT * 2) + part * 16;
      asm volatile("global_load_async_to_lds_b128 %0, %1, off" :: "v"(lds), "v"(gp) : "memory");
    }
    {
      const int c = tid + 256;
      const int row = c >> 2, part = c & 3;
      const ushort_t* gp = gbase + (long)(base + row) * ld + part * 8;
      const unsigned lds = lbase + row * (LDT * 2) + part * 16;
      asm volatile("global_load_async_to_lds_b128 %0, %1, off" :: "v"(lds), "v"(gp) : "memory");
    }
  };
  auto stage = [&](int buf, int k0) {
    async_tile(&sA[buf][0], A + k0, lda, rowBase);
    async_tile(&sB[buf][0], Bm + k0, ldb, colBase);
  };

  v8f acc[2][4];
#pragma unroll
  for (int mi = 0; mi < 2; ++mi)
#pragma unroll
    for (int ni = 0; ni < 4; ++ni)
#pragma unroll
      for (int r = 0; r < 8; ++r) acc[mi][ni][r] = 0.f;

  stage(0, 0);
  asm volatile("s_wait_asynccnt 0x0" ::: "memory");
  __syncthreads();

  int buf = 0;
  for (int k0 = 0; k0 < K; k0 += BK) {
    if (k0 + BK < K) stage(buf ^ 1, k0 + BK);    // overlap next tile with WMMA

    const int ka = lh * 8;   // ISA 16-bit layout: K chunks [ka,ka+7],[16+ka,...]
    v16bf afrag[2], bfrag[4];
#pragma unroll
    for (int mi = 0; mi < 2; ++mi) {
      const ushort_t* rp = &sA[buf][(wm * 32 + mi * 16 + lr) * LDT];
      v8bf lo = *reinterpret_cast<const v8bf*>(rp + ka);
      v8bf hi = *reinterpret_cast<const v8bf*>(rp + 16 + ka);
      afrag[mi] = __builtin_shufflevector(lo, hi, 0,1,2,3,4,5,6,7,8,9,10,11,12,13,14,15);
    }
#pragma unroll
    for (int ni = 0; ni < 4; ++ni) {
      const ushort_t* rp = &sB[buf][(wn * 64 + ni * 16 + lr) * LDT];
      v8bf lo = *reinterpret_cast<const v8bf*>(rp + ka);
      v8bf hi = *reinterpret_cast<const v8bf*>(rp + 16 + ka);
      bfrag[ni] = __builtin_shufflevector(lo, hi, 0,1,2,3,4,5,6,7,8,9,10,11,12,13,14,15);
    }
#pragma unroll
    for (int mi = 0; mi < 2; ++mi)
#pragma unroll
      for (int ni = 0; ni < 4; ++ni)
        acc[mi][ni] = __builtin_amdgcn_wmma_f32_16x16x32_bf16(
            false, afrag[mi], false, bfrag[ni], (short)0, acc[mi][ni], false, false);

    asm volatile("s_wait_asynccnt 0x0" ::: "memory");
    __syncthreads();
    buf ^= 1;
  }

  float* Cf = (float*)Cv;
  ushort_t* Cb = (ushort_t*)Cv;
#pragma unroll
  for (int mi = 0; mi < 2; ++mi)
#pragma unroll
    for (int ni = 0; ni < 4; ++ni)
#pragma unroll
      for (int r = 0; r < 8; ++r) {
        const int row = rowBase + wm * 32 + mi * 16 + lh * 8 + r;   // M mult of 128
        const int col = colBase + wn * 64 + ni * 16 + lr;
        if (col < Nstore) {
          float v = 0.f;
          if (col < N) {
            v = acc[mi][ni][r] + (bias ? bias[col] : 0.f);
            if (relu) v = fmaxf(v, 0.f);
          }
          const long idx = c_trans ? ((long)col * ldc + row) : ((long)row * ldc + col);
          if (c_bf16) Cb[idx] = f2bf(v);
          else        Cf[idx] = v;
        }
      }
}

// ---------------------------------------------------------------------------
// Conversion / packing kernels (f32 -> padded bf16, zero-filled pads)
// ---------------------------------------------------------------------------
__global__ void cvt_nk_bf16(const float* __restrict__ src, ushort_t* __restrict__ dst,
                            int N, int K, int Kpad) {   // grid.x = Npad
  const long n = blockIdx.x;
  const int k = blockIdx.y * 256 + threadIdx.x;
  if (k >= Kpad) return;
  const float v = (n < N && k < K) ? src[n * (long)K + k] : 0.f;
  dst[n * Kpad + k] = f2bf(v);
}
__global__ void cvt_kn_t_bf16(const float* __restrict__ src, ushort_t* __restrict__ dst,
                              int N, int K, int Kpad) { // src [K][N] -> dst [Npad][Kpad]
  const long n = blockIdx.x;
  const int k = blockIdx.y * 256 + threadIdx.x;
  if (k >= Kpad) return;
  const float v = (n < N && k < K) ? src[(long)k * N + n] : 0.f;
  dst[n * Kpad + k] = f2bf(v);
}
// head-pad Q/K weights: src [128][K] -> dst [256][K]; dst row 32h+j = src row
// 16h+j for j<16, zero rows otherwise (head stride 16 -> 32 with zero pad).
__global__ void cvt_headpad_bf16(const float* __restrict__ src, ushort_t* __restrict__ dst,
                                 int K) {               // grid (256, K/128), block 128
  const int n = blockIdx.x;
  const int k = blockIdx.y * 128 + threadIdx.x;
  if (k >= K) return;
  const int h = n >> 5, j = n & 31;
  const float v = (j < 16) ? src[(long)(h * 16 + j) * K + k] : 0.f;
  dst[(long)n * K + k] = f2bf(v);
}
__global__ void pad_bias(const float* __restrict__ b, float* __restrict__ o) { // block 256
  const int n = threadIdx.x;
  const int h = n >> 5, j = n & 31;
  o[n] = (j < 16) ? b[h * 16 + j] : 0.f;
}

// ---------------------------------------------------------------------------
// Graph / elementwise kernels
// ---------------------------------------------------------------------------
__global__ void fill_zero(float* p, long n) {
  long i = blockIdx.x * (long)blockDim.x + threadIdx.x;
  if (i < n) p[i] = 0.f;
}
__global__ void deg_kernel(const int* __restrict__ dst, float* __restrict__ deg, int nE) {
  int e = blockIdx.x * blockDim.x + threadIdx.x;
  if (e < nE) atomicAdd(&deg[dst[e]], 1.0f);
}
__global__ void dinv_kernel(const float* __restrict__ deg, float* __restrict__ dinv, int n) {
  int i = blockIdx.x * blockDim.x + threadIdx.x;
  if (i < n) dinv[i] = rsqrtf(deg[i] + 1.0f);
}
// acc[dst] += h[src]*dinv[src]*dinv[dst]   (grid: [edges, F/256]) ; h bf16
__global__ void msg_kernel(const ushort_t* __restrict__ h, int ldh,
                           const int* __restrict__ src, const int* __restrict__ dst,
                           const float* __restrict__ dinv, float* __restrict__ acc, int F) {
  const int e = blockIdx.x;
  const int f = blockIdx.y * 256 + threadIdx.x;
  if (f >= F) return;
  const int s = src[e], d = dst[e];
  const float w = dinv[s] * dinv[d];
  atomicAdd(&acc[(long)d * F + f], bf2f(h[(long)s * ldh + f]) * w);
}
// h = relu(acc + h*dinv^2 + bias)  in place over bf16 XW buffer
__global__ void gcn_finalize(const float* __restrict__ acc, ushort_t* __restrict__ h, int ldh,
                             const float* __restrict__ dinv, const float* __restrict__ bias, int F) {
  const long node = blockIdx.x;
  const int f = blockIdx.y * 256 + threadIdx.x;
  if (f >= F) return;
  const float di = dinv[node];
  const float v = acc[node * F + f] + bf2f(h[node * ldh + f]) * di * di + bias[f];
  h[node * ldh + f] = f2bf(fmaxf(v, 0.f));
}
// segment max into f32 (uint atomicMax valid: values >= 0, zero-initialized)
__global__ void segmax_kernel(const ushort_t* __restrict__ h, int ldh,
                              const int* __restrict__ batch, unsigned* __restrict__ g, int F) {
  const long node = blockIdx.x;
  const int f = blockIdx.y * 256 + threadIdx.x;
  if (f >= F) return;
  atomicMax(&g[(long)batch[node] * F + f], __float_as_uint(bf2f(h[node * ldh + f])));
}

// fused conv1d(k=8) + ReLU + maxpool3, bf16 in/out, weights async-staged to LDS
__global__ void conv_relu_pool(const ushort_t* __restrict__ in, const float* __restrict__ w,
                               const float* __restrict__ bias, ushort_t* __restrict__ out,
                               int Cin, int Lin, int Cout, int Lout) {
  __shared__ __align__(16) float sw[512];            // up to 64*8 weights
  const int b  = blockIdx.x;
  const int co = blockIdx.y;
  const int tid = threadIdx.x;
  const int nvec = Cin * 2;                          // 16B chunks of w[co]
  if (tid < nvec) {
    const unsigned lds = (unsigned)(size_t)(&sw[0]) + tid * 16;
    const float* gp = w + (long)co * Cin * 8 + tid * 4;
    asm volatile("global_load_async_to_lds_b128 %0, %1, off" :: "v"(lds), "v"(gp) : "memory");
  }
  asm volatile("s_wait_asynccnt 0x0" ::: "memory");
  __syncthreads();
  const int lp = blockIdx.z * blockDim.x + tid;
  if (lp >= Lout) return;
  const ushort_t* ip = in + (long)b * Cin * Lin;
  float m = -3.4e38f;
  for (int t = 0; t < 3; ++t) {
    const int pos = lp * 3 + t;
    float a = bias[co];
    for (int ci = 0; ci < Cin; ++ci) {
      const ushort_t* row = ip + (long)ci * Lin + pos;
      const float* wr = sw + ci * 8;
#pragma unroll
      for (int k = 0; k < 8; ++k) a += bf2f(row[k]) * wr[k];
    }
    m = fmaxf(m, fmaxf(a, 0.f));
  }
  out[((long)b * Cout + co) * Lout + lp] = f2bf(m);
}

// row softmax: f32 scores in -> bf16 probs out; 256-thread block per row
__global__ void softmax_rows(const float* __restrict__ s, ushort_t* __restrict__ o,
                             int L, float scale) {
  const float* p = s + (long)blockIdx.x * L;
  ushort_t* q = o + (long)blockIdx.x * L;
  const int tid = threadIdx.x;
  __shared__ float red[256];
  float mx = -3.4e38f;
  for (int i = tid; i < L; i += 256) mx = fmaxf(mx, p[i]);
  red[tid] = mx; __syncthreads();
  for (int st = 128; st > 0; st >>= 1) { if (tid < st) red[tid] = fmaxf(red[tid], red[tid + st]); __syncthreads(); }
  mx = red[0]; __syncthreads();
  float sum = 0.f;
  for (int i = tid; i < L; i += 256) { const float e = __expf((p[i] - mx) * scale); q[i] = f2bf(e); sum += e; }
  red[tid] = sum; __syncthreads();
  for (int st = 128; st > 0; st >>= 1) { if (tid < st) red[tid] += red[tid + st]; __syncthreads(); }
  const float inv = 1.f / red[0]; __syncthreads();
  for (int i = tid; i < L; i += 256) q[i] = f2bf(bf2f(q[i]) * inv);
}

// out = relu(LN(x + res) * gamma + beta) as bf16, strided; blockDim = 128
__global__ void add_ln_relu(const float* __restrict__ x, const ushort_t* __restrict__ res,
                            const float* __restrict__ gamma, const float* __restrict__ beta,
                            ushort_t* __restrict__ out, int ldo) {
  const int row = blockIdx.x, tid = threadIdx.x;
  __shared__ float red[DIM];
  const float v = x[(long)row * DIM + tid] + bf2f(res[(long)row * DIM + tid]);
  red[tid] = v; __syncthreads();
  for (int st = 64; st > 0; st >>= 1) { if (tid < st) red[tid] += red[tid + st]; __syncthreads(); }
  const float mean = red[0] * (1.f / DIM); __syncthreads();
  const float d = v - mean;
  red[tid] = d * d; __syncthreads();
  for (int st = 64; st > 0; st >>= 1) { if (tid < st) red[tid] += red[tid + st]; __syncthreads(); }
  const float inv = rsqrtf(red[0] * (1.f / DIM) + 1e-5f);
  out[(long)row * ldo + tid] = f2bf(fmaxf(d * inv * gamma[tid] + beta[tid], 0.f));
}

// out[b] = dot(comb2[b,:128], Wout) + bout ; blockDim = 128
__global__ void final_out(const float* __restrict__ c, const float* __restrict__ Wout,
                          const float* __restrict__ bout, float* __restrict__ out) {
  const int b = blockIdx.x, tid = threadIdx.x;
  __shared__ float red[DIM];
  red[tid] = c[(long)b * DIM + tid] * Wout[tid]; __syncthreads();
  for (int st = 64; st > 0; st >>= 1) { if (tid < st) red[tid] += red[tid + st]; __syncthreads(); }
  if (tid == 0) out[b] = red[0] + bout[0];
}

// ---------------------------------------------------------------------------
static inline int pad32(int x)  { return (x + 31) & ~31; }
static inline int pad128(int x) { return (x + 127) & ~127; }

extern "C" void kernel_launch(void* const* d_in, const int* in_sizes, int n_in,
                              void* d_out, int out_size, void* d_ws, size_t ws_size,
                              hipStream_t stream) {
  (void)in_sizes; (void)n_in; (void)out_size; (void)ws_size;
  const float* x      = (const float*)d_in[0];
  const float* cell_x = (const float*)d_in[1];
  const float *W1 = (const float*)d_in[2],  *b1 = (const float*)d_in[3];
  const float *W2 = (const float*)d_in[4],  *b2 = (const float*)d_in[5];
  const float *W3 = (const float*)d_in[6],  *b3 = (const float*)d_in[7];
  const float *Wg1 = (const float*)d_in[8], *bg1 = (const float*)d_in[9];
  const float *Wg2 = (const float*)d_in[10],*bg2 = (const float*)d_in[11];
  const float *Wc1 = (const float*)d_in[12],*bc1 = (const float*)d_in[13];
  const float *Wc2 = (const float*)d_in[14],*bc2 = (const float*)d_in[15];
  const float *Wc3 = (const float*)d_in[16],*bc3 = (const float*)d_in[17];
  const float *Wfc = (const float*)d_in[18],*bfc = (const float*)d_in[19];
  const float *Win1 = (const float*)d_in[20],*bin1 = (const float*)d_in[21];
  const float *Wo1  = (const float*)d_in[22],*bo1  = (const float*)d_in[23];
  const float *Win2 = (const float*)d_in[24],*bin2 = (const float*)d_in[25];
  const float *Wo2  = (const float*)d_in[26],*bo2  = (const float*)d_in[27];
  const float *g1 = (const float*)d_in[28], *be1 = (const float*)d_in[29];
  const float *g2 = (const float*)d_in[30], *be2 = (const float*)d_in[31];
  const float *Wcomb = (const float*)d_in[32],*bcomb = (const float*)d_in[33];
  const float *Wout  = (const float*)d_in[34],*bout  = (const float*)d_in[35];
  const int* edge  = (const int*)d_in[36];
  const int* batch = (const int*)d_in[37];
  const int* esrc = edge;
  const int* edst = edge + EDGES;

  const int F1 = FD, F2 = 2 * FD, F4 = 4 * FD;                   // 334/668/1336
  const int F1p = pad32(F1), F2p = pad32(F2), F4p = pad32(F4);   // 352/672/1344

  // ---- workspace carving
  size_t ofs = 0;
  char* wsb = (char*)d_ws;
  auto carve = [&](size_t bytes) -> void* {
    void* p = wsb + ofs; ofs += (bytes + 255) & ~(size_t)255; return p;
  };
  float*    accF  = (float*)carve((size_t)NODES * F4 * 4);        // scatter acc; alias: f32 scores
  ushort_t* hBF1  = (ushort_t*)carve((size_t)NODES * F4p * 2);    // bf16 ping (h1,h3)
  ushort_t* hBF2  = (ushort_t*)carve((size_t)NODES * F4p * 2);    // bf16 pong (h2); alias: x_bf/probs/CNN
  float*    deg   = (float*)carve((size_t)NODES * 4);
  float*    dinv  = (float*)carve((size_t)NODES * 4);
  float*    gpool = (float*)carve((size_t)BATCH * F4 * 4);
  ushort_t* gpoolB= (ushort_t*)carve((size_t)BATCH * F4p * 2);
  ushort_t* gh    = (ushort_t*)carve((size_t)BATCH * 1024 * 2);
  ushort_t* gvec  = (ushort_t*)carve((size_t)BATCH * DIM * 2);
  ushort_t* xt    = (ushort_t*)carve((size_t)BATCH * DIM * 2);
  ushort_t* bufQp = (ushort_t*)carve((size_t)BATCH * 2 * DIM * 2);   // [2048,256] head-padded
  ushort_t* bufKp = (ushort_t*)carve((size_t)BATCH * 2 * DIM * 2);   // [2048,256] head-padded
  ushort_t* bufVt = (ushort_t*)carve((size_t)256 * BATCH * 2);       // [256,2048] (128 valid + OOB pad)
  ushort_t* bufO  = (ushort_t*)carve((size_t)BATCH * DIM * 2);
  float*    bufP  = (float*)carve((size_t)BATCH * DIM * 4);
  ushort_t* comb  = (ushort_t*)carve((size_t)BATCH * 2 * DIM * 2);
  float*    comb2 = (float*)carve((size_t)BATCH * DIM * 4);
  ushort_t* cellB = (ushort_t*)carve((size_t)BATCH * LCELL * 2);
  // bf16 weight scratch ([Npad][Kpad], zero-padded)
  ushort_t* W1t  = (ushort_t*)carve((size_t)pad128(F1) * F1p * 2);
  ushort_t* W2t  = (ushort_t*)carve((size_t)pad128(F2) * F1p * 2);
  ushort_t* W3t  = (ushort_t*)carve((size_t)pad128(F4) * F2p * 2);
  ushort_t* Wg1t = (ushort_t*)carve((size_t)1024 * F4p * 2);
  ushort_t* Wg2t = (ushort_t*)carve((size_t)DIM * 1024 * 2);
  ushort_t* Wfct = (ushort_t*)carve((size_t)DIM * 4096 * 2);
  ushort_t* Wq1p = (ushort_t*)carve((size_t)2 * DIM * DIM * 2);   // [256,128] head-padded
  ushort_t* Wk1p = (ushort_t*)carve((size_t)2 * DIM * DIM * 2);
  ushort_t* Wq2p = (ushort_t*)carve((size_t)2 * DIM * DIM * 2);
  ushort_t* Wk2p = (ushort_t*)carve((size_t)2 * DIM * DIM * 2);
  ushort_t* Wv1t = (ushort_t*)carve((size_t)DIM * DIM * 2);
  ushort_t* Wv2t = (ushort_t*)carve((size_t)DIM * DIM * 2);
  ushort_t* Wo1t = (ushort_t*)carve((size_t)DIM * DIM * 2);
  ushort_t* Wo2t = (ushort_t*)carve((size_t)DIM * DIM * 2);
  ushort_t* Wcmbt= (ushort_t*)carve((size_t)DIM * 2 * DIM * 2);
  float*    bq1p = (float*)carve(256 * 4);
  float*    bk1p = (float*)carve(256 * 4);
  float*    bq2p = (float*)carve(256 * 4);
  float*    bk2p = (float*)carve(256 * 4);
  // aliases into the big buffers (phases don't overlap in stream order)
  ushort_t* x_bf   = hBF2;                                   // phase 1 only
  ushort_t* probsB = hBF2;                                   // attention phase
  ushort_t* c1p = hBF2 + (size_t)50 * 1024 * 1024;           // CNN phase
  ushort_t* c2p = c1p + (size_t)BATCH * 32 * 316;
  ushort_t* c3p = c2p + (size_t)BATCH * 64 * 103;            // [B,128,32] -> flat 4096
  float*    scoresF = accF;                                  // 8*2048*2048*4 = 134MB

  auto gemm = [&](const ushort_t* A, long lda, const ushort_t* B, long ldb,
                  const float* bias, void* C, long ldc, int c_bf16, int c_trans,
                  int M, int N, int Nstore, int K, int relu) {
    dim3 grid((Nstore + BN - 1) / BN, M / BM);
    gemm_bf16_wmma<<<grid, 256, 0, stream>>>(A, lda, B, ldb, bias, C, ldc, c_bf16, c_trans,
                                             M, N, Nstore, K, relu);
  };
  auto cnk = [&](const float* s, ushort_t* d, int N, int K, int Kpad, int Npad) {
    dim3 g(Npad, (Kpad + 255) / 256);
    cvt_nk_bf16<<<g, 256, 0, stream>>>(s, d, N, K, Kpad);
  };
  auto cknt = [&](const float* s, ushort_t* d, int N, int K, int Kpad, int Npad) {
    dim3 g(Npad, (Kpad + 255) / 256);
    cvt_kn_t_bf16<<<g, 256, 0, stream>>>(s, d, N, K, Kpad);
  };

  // ---- weight + input conversions to padded bf16
  cnk (x, x_bf, NODES, F1, F1p, NODES);
  cknt(W1, W1t, F1, F1, F1p, pad128(F1));
  cknt(W2, W2t, F2, F1, F1p, pad128(F2));
  cknt(W3, W3t, F4, F2, F2p, pad128(F4));
  cnk (Wg1, Wg1t, 1024, F4, F4p, 1024);
  cnk (Wg2, Wg2t, DIM, 1024, 1024, DIM);
  cnk (Wfc, Wfct, DIM, 4096, 4096, DIM);
  { dim3 g(2 * DIM, 1);
    cvt_headpad_bf16<<<g, 128, 0, stream>>>(Win1,             Wq1p, DIM);
    cvt_headpad_bf16<<<g, 128, 0, stream>>>(Win1 + DIM * DIM, Wk1p, DIM);
    cvt_headpad_bf16<<<g, 128, 0, stream>>>(Win2,             Wq2p, DIM);
    cvt_headpad_bf16<<<g, 128, 0, stream>>>(Win2 + DIM * DIM, Wk2p, DIM); }
  pad_bias<<<1, 256, 0, stream>>>(bin1,       bq1p);
  pad_bias<<<1, 256, 0, stream>>>(bin1 + DIM, bk1p);
  pad_bias<<<1, 256, 0, stream>>>(bin2,       bq2p);
  pad_bias<<<1, 256, 0, stream>>>(bin2 + DIM, bk2p);
  cnk (Win1 + 2 * DIM * DIM, Wv1t, DIM, DIM, DIM, DIM);
  cnk (Win2 + 2 * DIM * DIM, Wv2t, DIM, DIM, DIM, DIM);
  cnk (Wo1, Wo1t, DIM, DIM, DIM, DIM);
  cnk (Wo2, Wo2t, DIM, DIM, DIM, DIM);
  cnk (Wcomb, Wcmbt, DIM, 2 * DIM, 2 * DIM, DIM);
  cnk (cell_x, cellB, BATCH, LCELL, LCELL, BATCH);

  // ---- graph normalization (graph fixed across layers)
  fill_zero<<<(NODES + 255) / 256, 256, 0, stream>>>(deg, NODES);
  deg_kernel<<<(EDGES + 255) / 256, 256, 0, stream>>>(edst, deg, EDGES);
  dinv_kernel<<<(NODES + 255) / 256, 256, 0, stream>>>(deg, dinv, NODES);

  // ---- GCN layers: XW (WMMA) -> zero acc -> edge scatter -> finalize (bf16 in place)
  auto gcn_layer = [&](const ushort_t* Xin, int Kp, const ushort_t* Wt,
                       const float* b, int Fout, int Fp, ushort_t* Hout) {
    gemm(Xin, Kp, Wt, Kp, nullptr, Hout, Fp, 1, 0, NODES, Fout, Fp, Kp, 0);
    const long tot = (long)NODES * Fout;
    fill_zero<<<(unsigned)((tot + 255) / 256), 256, 0, stream>>>(accF, tot);
    dim3 mg(EDGES, (Fout + 255) / 256);
    msg_kernel<<<mg, 256, 0, stream>>>(Hout, Fp, esrc, edst, dinv, accF, Fout);
    dim3 fg(NODES, (Fout + 255) / 256);
    gcn_finalize<<<fg, 256, 0, stream>>>(accF, Hout, Fp, dinv, b, Fout);
  };
  gcn_layer(x_bf, F1p, W1t, b1, F1, F1p, hBF1);   // h1 -> hBF1
  gcn_layer(hBF1, F1p, W2t, b2, F2, F2p, hBF2);   // h2 -> hBF2 (x_bf dead)
  gcn_layer(hBF2, F2p, W3t, b3, F4, F4p, hBF1);   // h3 -> hBF1

  // ---- global max pool + drug MLP
  fill_zero<<<(unsigned)(((long)BATCH * F4 + 255) / 256), 256, 0, stream>>>(gpool, (long)BATCH * F4);
  { dim3 sg(NODES, (F4 + 255) / 256);
    segmax_kernel<<<sg, 256, 0, stream>>>(hBF1, F4p, batch, (unsigned*)gpool, F4); }
  cnk(gpool, gpoolB, BATCH, F4, F4p, BATCH);
  gemm(gpoolB, F4p, Wg1t, F4p, bg1, gh, 1024, 1, 0, BATCH, 1024, 1024, F4p, 1);
  gemm(gh, 1024, Wg2t, 1024, bg2, gvec, DIM, 1, 0, BATCH, DIM, DIM, 1024, 0);

  // ---- CNN branch (fused conv+relu+pool3; 955->316->103->32), then Wfc
  { dim3 g(BATCH, 32, (316 + 127) / 128);
    conv_relu_pool<<<g, 128, 0, stream>>>(cellB, Wc1, bc1, c1p, 1, LCELL, 32, 316); }
  { dim3 g(BATCH, 64, 1);
    conv_relu_pool<<<g, 128, 0, stream>>>(c1p, Wc2, bc2, c2p, 32, 316, 64, 103); }
  { dim3 g(BATCH, 128, 1);
    conv_relu_pool<<<g, 128, 0, stream>>>(c2p, Wc3, bc3, c3p, 64, 103, 128, 32); }
  gemm(c3p, 4096, Wfct, 4096, bfc, xt, DIM, 1, 0, BATCH, DIM, DIM, 4096, 0);

  // ---- cross-attention blocks, each writing its half of comb (relu folded in LN)
  auto mha_block = [&](const ushort_t* srcQ, const ushort_t* srcKV,
                       const ushort_t* Wqp, const float* bqp,
                       const ushort_t* Wkp, const float* bkp,
                       const ushort_t* Wv, const float* bv,
                       const ushort_t* Wot, const float* bo,
                       const float* gam, const float* bet, ushort_t* outp) {
    // head-padded Q/K [2048, 8*32] (pad cols exactly zero), transposed V [128,2048]
    gemm(srcQ,  DIM, Wqp, DIM, bqp, bufQp, 2 * DIM, 1, 0, BATCH, 2 * DIM, 2 * DIM, DIM, 0);
    gemm(srcKV, DIM, Wkp, DIM, bkp, bufKp, 2 * DIM, 1, 0, BATCH, 2 * DIM, 2 * DIM, DIM, 0);
    gemm(srcKV, DIM, Wv,  DIM, bv,  bufVt, BATCH, 1, 1, BATCH, DIM, DIM, DIM, 0);
    for (int h = 0; h < HEADS; ++h)   // scores_h = Q_h @ K_h^T (K=32, zero-padded heads)
      gemm(bufQp + 32 * h, 2 * DIM, bufKp + 32 * h, 2 * DIM, nullptr,
           scoresF + (long)h * BATCH * BATCH, BATCH, 0, 0, BATCH, BATCH, BATCH, 32, 0);
    softmax_rows<<<HEADS * BATCH, 256, 0, stream>>>(scoresF, probsB, BATCH, 0.25f);
    for (int h = 0; h < HEADS; ++h)   // O_h = P_h @ Vt_h^T (N=16, K=2048)
      gemm(probsB + (long)h * BATCH * BATCH, BATCH, bufVt + (long)h * 16 * BATCH, BATCH,
           nullptr, bufO + 16 * h, DIM, 1, 0, BATCH, 16, 16, BATCH, 0);
    gemm(bufO, DIM, Wot, DIM, bo, bufP, DIM, 0, 0, BATCH, DIM, DIM, DIM, 0);
    add_ln_relu<<<BATCH, DIM, 0, stream>>>(bufP, srcQ, gam, bet, outp, 2 * DIM);
  };
  mha_block(gvec, xt, Wq1p, bq1p, Wk1p, bk1p, Wv1t, bin1 + 2 * DIM,
            Wo1t, bo1, g1, be1, comb);                               // -> comb[:, :128]
  mha_block(xt, gvec, Wq2p, bq2p, Wk2p, bk2p, Wv2t, bin2 + 2 * DIM,
            Wo2t, bo2, g2, be2, comb + DIM);                         // -> comb[:, 128:]

  // ---- head
  gemm(comb, 2 * DIM, Wcmbt, 2 * DIM, bcomb, comb2, DIM, 0, 0, BATCH, DIM, DIM, 2 * DIM, 1);
  final_out<<<BATCH, DIM, 0, stream>>>(comb2, Wout, bout, (float*)d_out);
}
